// AttentionBlock_4561255269140
// MI455X (gfx1250) — compile-verified, matrix operand order checked by reading, not run on hardware
//
#include <hip/hip_runtime.h>
#include <math.h>

// ---------------- CDNA5 / gfx1250 attention block ----------------
// bf16 WMMA for all GEMMs + flash attention with async-LDS K/V staging.

typedef __bf16 bf16;
typedef __attribute__((ext_vector_type(16))) __bf16 v16bf;
typedef __attribute__((ext_vector_type(8)))  __bf16 v8bf;
typedef __attribute__((ext_vector_type(8)))  float  v8f;

#define B_    4
#define C_    256
#define HW_   4096
#define G_    32
#define CPG   8
#define NH    4
#define DK    64
#define M_TOT (B_*HW_)     // 16384
#define QKV_N 768
#define EPS_  1e-5f
#define SCALE_ 0.35355339059327379f   // 64^(-1/4)
#define INV_SQRT2_ 0.70710678118654752f

// Load a 16x32 bf16 A/B fragment row-chunk for this lane: elements 0..7 from
// p0 (K = kA..kA+7), elements 8..15 from p1 (K = kA+16..kA+23).
__device__ __forceinline__ v16bf load_ab(const bf16* p0, const bf16* p1) {
  v8bf lo = *(const v8bf*)p0;
  v8bf hi = *(const v8bf*)p1;
  v16bf r;
#pragma unroll
  for (int i = 0; i < 8; ++i) { r[i] = lo[i]; r[i + 8] = hi[i]; }
  return r;
}

__device__ __forceinline__ v8f wmma_bf16(v16bf a, v16bf b, v8f c) {
  return __builtin_amdgcn_wmma_f32_16x16x32_bf16(false, a, false, b, (short)0, c,
                                                 false, false);
}

// Async 16-byte global->LDS copy (ASYNCcnt-tracked, gfx1250).
// Low 32 bits of a generic pointer to __shared__ == LDS allocation offset.
__device__ __forceinline__ void async_cp16(const bf16* g, const bf16* l) {
  unsigned lo = (unsigned)(uintptr_t)l;
  asm volatile("global_load_async_to_lds_b128 %0, %1, off"
               :: "v"(lo), "v"((unsigned long long)(uintptr_t)g)
               : "memory");
}

// ---------------- 1. GroupNorm statistics ----------------
__global__ void gn_stats(const float* __restrict__ x, float* __restrict__ mean,
                         float* __restrict__ rstd) {
  int bg = blockIdx.x;  // b*32 + g, groups are 8 consecutive channels
  const float* p = x + (size_t)bg * CPG * HW_;
  float s = 0.f, sq = 0.f;
  for (int i = threadIdx.x; i < CPG * HW_; i += 256) {
    float v = p[i];
    s += v; sq += v * v;
  }
  __shared__ float sh1[256], sh2[256];
  sh1[threadIdx.x] = s; sh2[threadIdx.x] = sq;
  __syncthreads();
  for (int o = 128; o > 0; o >>= 1) {
    if ((int)threadIdx.x < o) {
      sh1[threadIdx.x] += sh1[threadIdx.x + o];
      sh2[threadIdx.x] += sh2[threadIdx.x + o];
    }
    __syncthreads();
  }
  if (threadIdx.x == 0) {
    float m = sh1[0] * (1.0f / (CPG * HW_));
    float v = sh2[0] * (1.0f / (CPG * HW_)) - m * m;
    mean[bg] = m;
    rstd[bg] = rsqrtf(v + EPS_);
  }
}

// ---------------- 2. Apply GN, write hid [B*N][C] bf16 ----------------
__global__ void gn_apply(const float* __restrict__ x, const float* __restrict__ mean,
                         const float* __restrict__ rstd, const float* __restrict__ gamma,
                         const float* __restrict__ beta, bf16* __restrict__ hid) {
  int bc = blockIdx.x;            // b*256 + c
  int b = bc >> 8, c = bc & 255;
  int bg = b * G_ + (c >> 3);
  float ga = gamma[c] * rstd[bg];
  float be = beta[c] - mean[bg] * ga;
  const float* xp = x + (size_t)bc * HW_;
  bf16* hp = hid + (size_t)b * HW_ * C_ + c;
  for (int n = threadIdx.x; n < HW_; n += 256)
    hp[(size_t)n * C_] = (bf16)(xp[n] * ga + be);
}

// ---------------- 3. Transpose weights to [Nout][K] bf16 ----------------
__global__ void prep_weights(const float* __restrict__ Wqkv, const float* __restrict__ Wout,
                             bf16* __restrict__ wqkvT, bf16* __restrict__ woutT) {
  int i = blockIdx.x * 256 + threadIdx.x;
  if (i < C_ * QKV_N) {
    int k = i / QKV_N, j = i % QKV_N;
    wqkvT[(size_t)j * C_ + k] = (bf16)Wqkv[i];
  }
  if (i < C_ * C_) {
    int k = i >> 8, j = i & 255;
    woutT[(size_t)j * C_ + k] = (bf16)Wout[i];
  }
}

// ---------------- 4. QKV projection GEMM (WMMA bf16) ----------------
// grid (48, 128), 256 threads = 8 waves; wave -> one 16x16 tile.
__global__ void __launch_bounds__(256) qkv_gemm(
    const bf16* __restrict__ hid, const bf16* __restrict__ wqkvT,
    const float* __restrict__ bqkv, bf16* __restrict__ Q, bf16* __restrict__ K,
    bf16* __restrict__ Vt) {
  int wave = threadIdx.x >> 5, lane = threadIdx.x & 31;
  int lh = lane & 15, hiHalf = lane >> 4, kA = hiHalf * 8;
  int mt = blockIdx.y * 8 + wave;
  int nt = blockIdx.x;
  const bf16* ap = hid + ((size_t)mt * 16 + lh) * C_;
  const bf16* bp = wqkvT + ((size_t)nt * 16 + lh) * C_;
  v8f acc = {};
#pragma unroll
  for (int kb = 0; kb < 8; ++kb) {
    int k0 = kb * 32 + kA;
    v16bf a = load_ab(ap + k0, ap + k0 + 16);
    v16bf b = load_ab(bp + k0, bp + k0 + 16);
    acc = wmma_bf16(a, b, acc);
  }
  int j = nt * 16 + lh;
  float bias = bqkv[j];
  int head = j / 192, r = j % 192;
  float s = (r < 128) ? SCALE_ : 1.0f;   // fold d_k^-1/4 into Q and K
#pragma unroll
  for (int e = 0; e < 8; ++e) {
    int grow = mt * 16 + e + hiHalf * 8;        // global row in [0, 16384)
    int bidx = grow >> 12, n = grow & 4095;
    bf16 bv = (bf16)((acc[e] + bias) * s);
    if (r < 64)
      Q[(((size_t)(bidx * NH + head)) * HW_ + n) * DK + r] = bv;
    else if (r < 128)
      K[(((size_t)(bidx * NH + head)) * HW_ + n) * DK + (r - 64)] = bv;
    else
      Vt[(((size_t)(bidx * NH + head)) * DK + (r - 128)) * HW_ + n] = bv;
  }
}

// ---------------- 5. Flash attention ----------------
// grid (32, NH, B), 256 threads = 8 waves; each wave: 16 queries x d=64.
// K/V chunks staged once per block via async global->LDS copies (double
// buffered, ASYNCcnt), WMMA fragments then read from LDS (8x less VMEM).
__global__ void __launch_bounds__(256) flash_attn(
    const bf16* __restrict__ Q, const bf16* __restrict__ K,
    const bf16* __restrict__ Vt, bf16* __restrict__ attnout) {
  __shared__ __align__(16) bf16 kbuf[2][32][DK];   // [buf][key][d]     4 KB ea
  __shared__ __align__(16) bf16 vbuf[2][DK][32];   // [buf][d][key]     4 KB ea
  __shared__ float plds[8][16][32];                // per-wave P staging
  int t = threadIdx.x;
  int wave = t >> 5, lane = t & 31;
  int lh = lane & 15, hiHalf = lane >> 4, kA = hiHalf * 8;
  int b = blockIdx.z, h = blockIdx.y;
  int q0 = blockIdx.x * 128 + wave * 16;
  const bf16* Qbase = Q + ((size_t)(b * NH + h)) * HW_ * DK;
  const bf16* Kbase = K + ((size_t)(b * NH + h)) * HW_ * DK;
  const bf16* Vbase = Vt + ((size_t)(b * NH + h)) * DK * HW_;

  // Q fragment stays in registers for the whole pass (16 x 64 bf16).
  const bf16* qr = Qbase + (size_t)(q0 + lh) * DK;
  v16bf qa0 = load_ab(qr + kA, qr + kA + 16);
  v16bf qa1 = load_ab(qr + 32 + kA, qr + 32 + kA + 16);

  v8f o0 = {}, o1 = {}, o2 = {}, o3 = {};
  float mrun[8], lrun[8];
#pragma unroll
  for (int e = 0; e < 8; ++e) { mrun[e] = -1e30f; lrun[e] = 0.f; }
  float (*pw)[32] = plds[wave];

  // Per-chunk copy assignment: thread t moves one 16B piece of K and of V.
  int krow = t >> 3, kcol = (t & 7) * 8;   // K: [32][64]
  int vrow = t >> 2, vcol = (t & 3) * 8;   // V: [64][32]

  // Prologue: stage chunk 0 into buffer 0.
  async_cp16(Kbase + (size_t)krow * DK + kcol, &kbuf[0][krow][kcol]);
  async_cp16(Vbase + (size_t)vrow * HW_ + vcol, &vbuf[0][vrow][vcol]);

  for (int j0 = 0; j0 < HW_; j0 += 32) {
    int cur = (j0 >> 5) & 1;
    if (j0 + 32 < HW_) {
      int nb = cur ^ 1;
      async_cp16(Kbase + (size_t)(j0 + 32 + krow) * DK + kcol, &kbuf[nb][krow][kcol]);
      async_cp16(Vbase + (size_t)vrow * HW_ + (j0 + 32) + vcol, &vbuf[nb][vrow][vcol]);
      asm volatile("s_wait_asynccnt 0x2" ::: "memory");  // chunk j0 landed
    } else {
      asm volatile("s_wait_asynccnt 0x0" ::: "memory");
    }
    __syncthreads();   // publish LDS chunk to all waves

    // S = Q K^T  (two 16x16 score tiles, contraction over d in 2 steps)
    v8f s0 = {}, s1 = {};
    {
      const bf16* kr0 = &kbuf[cur][lh][0];
      const bf16* kr1 = &kbuf[cur][16 + lh][0];
      v16bf kb;
      kb = load_ab(kr0 + kA, kr0 + kA + 16);
      s0 = wmma_bf16(qa0, kb, s0);
      kb = load_ab(kr0 + 32 + kA, kr0 + 32 + kA + 16);
      s0 = wmma_bf16(qa1, kb, s0);
      kb = load_ab(kr1 + kA, kr1 + kA + 16);
      s1 = wmma_bf16(qa0, kb, s1);
      kb = load_ab(kr1 + 32 + kA, kr1 + 32 + kA + 16);
      s1 = wmma_bf16(qa1, kb, s1);
    }
    // Online softmax. C-layout: element e = row (e + 8*hiHalf), lanes = 16 cols.
    float sc[8];
#pragma unroll
    for (int e = 0; e < 8; ++e) {
      float tm = fmaxf(s0[e], s1[e]);
      tm = fmaxf(tm, __shfl_xor(tm, 1, 32));
      tm = fmaxf(tm, __shfl_xor(tm, 2, 32));
      tm = fmaxf(tm, __shfl_xor(tm, 4, 32));
      tm = fmaxf(tm, __shfl_xor(tm, 8, 32));
      float nm = fmaxf(mrun[e], tm);
      float so = __expf(mrun[e] - nm);
      float p0 = __expf(s0[e] - nm);
      float p1 = __expf(s1[e] - nm);
      s0[e] = p0; s1[e] = p1;
      float rs = p0 + p1;
      rs += __shfl_xor(rs, 1, 32);
      rs += __shfl_xor(rs, 2, 32);
      rs += __shfl_xor(rs, 4, 32);
      rs += __shfl_xor(rs, 8, 32);
      lrun[e] = lrun[e] * so + rs;
      mrun[e] = nm;
      sc[e] = so;
    }
#pragma unroll
    for (int e = 0; e < 8; ++e) {
      o0[e] *= sc[e]; o1[e] *= sc[e]; o2[e] *= sc[e]; o3[e] *= sc[e];
    }
    // Re-layout P: C-layout -> row-major LDS -> A-layout bf16 fragment.
#pragma unroll
    for (int e = 0; e < 8; ++e) {
      pw[e + hiHalf * 8][lh]      = s0[e];
      pw[e + hiHalf * 8][16 + lh] = s1[e];
    }
    asm volatile("s_wait_dscnt 0" ::: "memory");   // cross-lane LDS RAW
    v16bf pa{};
    {
      const float* prow = pw[lh];
#pragma unroll
      for (int i = 0; i < 8; ++i) {
        pa[i]     = (bf16)prow[kA + i];
        pa[i + 8] = (bf16)prow[kA + 16 + i];
      }
    }
    asm volatile("s_wait_dscnt 0" ::: "memory");   // WAR vs next iter's stores
    // O += P V  (V in LDS, transposed: lane's column d is contiguous in keys)
    {
      const bf16* vr = &vbuf[cur][lh][0];
      v16bf vb;
      vb = load_ab(vr + kA, vr + kA + 16);
      o0 = wmma_bf16(pa, vb, o0);
      vr += 16 * 32;
      vb = load_ab(vr + kA, vr + kA + 16);
      o1 = wmma_bf16(pa, vb, o1);
      vr += 16 * 32;
      vb = load_ab(vr + kA, vr + kA + 16);
      o2 = wmma_bf16(pa, vb, o2);
      vr += 16 * 32;
      vb = load_ab(vr + kA, vr + kA + 16);
      o3 = wmma_bf16(pa, vb, o3);
    }
    __syncthreads();   // all waves done reading before buffer reuse
  }
  // Normalize and store head output into attnout [B*N][C] (head-concat cols).
#pragma unroll
  for (int e = 0; e < 8; ++e) {
    float inv = 1.0f / lrun[e];
    int row = q0 + e + hiHalf * 8;
    size_t base = ((size_t)(b * HW_ + row)) * C_ + h * DK + lh;
    attnout[base]      = (bf16)(o0[e] * inv);
    attnout[base + 16] = (bf16)(o1[e] * inv);
    attnout[base + 32] = (bf16)(o2[e] * inv);
    attnout[base + 48] = (bf16)(o3[e] * inv);
  }
}

// ---------------- 6. Output projection + residual (WMMA bf16) ----------------
// grid (16, 128), 256 threads = 8 waves.
__global__ void __launch_bounds__(256) out_gemm(
    const bf16* __restrict__ attnout, const bf16* __restrict__ woutT,
    const float* __restrict__ bout, const float* __restrict__ x,
    float* __restrict__ out) {
  int wave = threadIdx.x >> 5, lane = threadIdx.x & 31;
  int lh = lane & 15, hiHalf = lane >> 4, kA = hiHalf * 8;
  int mt = blockIdx.y * 8 + wave;
  int nt = blockIdx.x;
  const bf16* ap = attnout + ((size_t)mt * 16 + lh) * C_;
  const bf16* bp = woutT + ((size_t)nt * 16 + lh) * C_;
  v8f acc = {};
#pragma unroll
  for (int kb = 0; kb < 8; ++kb) {
    int k0 = kb * 32 + kA;
    v16bf a = load_ab(ap + k0, ap + k0 + 16);
    v16bf b = load_ab(bp + k0, bp + k0 + 16);
    acc = wmma_bf16(a, b, acc);
  }
  int c = nt * 16 + lh;
  float bias = bout[c];
#pragma unroll
  for (int e = 0; e < 8; ++e) {
    int grow = mt * 16 + e + hiHalf * 8;
    int bidx = grow >> 12, n = grow & 4095;
    size_t xo = ((size_t)bidx * C_ + c) * HW_ + n;   // output is [B][C][H*W]
    out[xo] = (acc[e] + bias + x[xo]) * INV_SQRT2_;
  }
}

// ---------------- launcher ----------------
extern "C" void kernel_launch(void* const* d_in, const int* in_sizes, int n_in,
                              void* d_out, int out_size, void* d_ws, size_t ws_size,
                              hipStream_t stream) {
  const float* x     = (const float*)d_in[0];
  const float* gamma = (const float*)d_in[1];
  const float* beta  = (const float*)d_in[2];
  const float* Wqkv  = (const float*)d_in[3];
  const float* bqkv  = (const float*)d_in[4];
  const float* Wout  = (const float*)d_in[5];
  const float* bout  = (const float*)d_in[6];
  float* out = (float*)d_out;

  char* ws = (char*)d_ws;
  float* mean = (float*)ws;
  float* rstd = mean + 256;
  size_t off = 4096;
  bf16* hid   = (bf16*)(ws + off); off += (size_t)M_TOT * C_ * 2;       // 8 MB
  bf16* wqkvT = (bf16*)(ws + off); off += (size_t)QKV_N * C_ * 2;
  bf16* woutT = (bf16*)(ws + off); off += (size_t)C_ * C_ * 2;
  bf16* Qb    = (bf16*)(ws + off); off += (size_t)B_ * NH * HW_ * DK * 2;
  bf16* Kb    = (bf16*)(ws + off); off += (size_t)B_ * NH * HW_ * DK * 2;
  bf16* Vt    = (bf16*)(ws + off); off += (size_t)B_ * NH * HW_ * DK * 2;
  bf16* attno = (bf16*)(ws + off); off += (size_t)M_TOT * C_ * 2;

  gn_stats<<<B_ * G_, 256, 0, stream>>>(x, mean, rstd);
  gn_apply<<<B_ * C_, 256, 0, stream>>>(x, mean, rstd, gamma, beta, hid);
  prep_weights<<<(C_ * QKV_N + 255) / 256, 256, 0, stream>>>(Wqkv, Wout, wqkvT, woutT);
  qkv_gemm<<<dim3(QKV_N / 16, M_TOT / 128), 256, 0, stream>>>(hid, wqkvT, bqkv, Qb, Kb, Vt);
  flash_attn<<<dim3(HW_ / 128, NH, B_), 256, 0, stream>>>(Qb, Kb, Vt, attno);
  out_gemm<<<dim3(C_ / 16, M_TOT / 128), 256, 0, stream>>>(attno, woutT, bout, x, out);
}
